// overAll_37606733644133
// MI455X (gfx1250) — compile-verified, matrix operand order checked by reading, not run on hardware
//
#include <hip/hip_runtime.h>

#define NN 50000
#define NE 800000
#define NR 2000
#define FF 128
#define FD 384
#define PP 64
#define EPSV 1e-12f

typedef __attribute__((ext_vector_type(16))) _Float16 v16h;
typedef __attribute__((ext_vector_type(8)))  float   v8f;

// ---------- wave32 helpers ----------
__device__ __forceinline__ float wsum(float v) {
#pragma unroll
  for (int o = 16; o > 0; o >>= 1) v += __shfl_xor(v, o, 32);
  return v;
}
__device__ __forceinline__ float wmaxr(float v) {
#pragma unroll
  for (int o = 16; o > 0; o >>= 1) v = fmaxf(v, __shfl_xor(v, o, 32));
  return v;
}
// monotonic float <-> uint encoding for atomicMax on floats
__device__ __forceinline__ unsigned encf(float f) {
  unsigned u = __float_as_uint(f);
  return (u & 0x80000000u) ? ~u : (u | 0x80000000u);
}
__device__ __forceinline__ float decf(unsigned e) {
  unsigned u = (e & 0x80000000u) ? (e & 0x7fffffffu) : ~e;
  return __uint_as_float(u);
}

// ---------- WMMA fragment helpers (16x16x32 f16, wave32 layouts) ----------
// A (16xK f32 row-major) -> 16x32 f16 fragment (no scale)
__device__ __forceinline__ v16h ldA(const float* A, int lda, int m0, int k0) {
  int lane = threadIdx.x & 31;
  int m = m0 + (lane & 15);
  int koff = (lane >> 4) * 8;           // lanes 16..31 hold K+8
  const float* row = A + (size_t)m * lda + k0;
  v16h a;
#pragma unroll
  for (int p = 0; p < 8; ++p) {
    int k = ((p < 4) ? 2 * p : 2 * p + 8) + koff;  // K {0..7,16..23} (+8 hi half)
    a[2 * p]     = (_Float16)row[k];
    a[2 * p + 1] = (_Float16)row[k + 1];
  }
  return a;
}
// A with per-row scale
__device__ __forceinline__ v16h ldAs(const float* A, int lda, int m0, int k0,
                                     const float* scale) {
  int lane = threadIdx.x & 31;
  int m = m0 + (lane & 15);
  int koff = (lane >> 4) * 8;
  float s = scale[m];
  const float* row = A + (size_t)m * lda + k0;
  v16h a;
#pragma unroll
  for (int p = 0; p < 8; ++p) {
    int k = ((p < 4) ? 2 * p : 2 * p + 8) + koff;
    a[2 * p]     = (_Float16)(row[k] * s);
    a[2 * p + 1] = (_Float16)(row[k + 1] * s);
  }
  return a;
}
// pre-packed fragment load: one aligned v16h (32B/lane) per (ktile, ntile)
__device__ __forceinline__ v16h ldP(const _Float16* packed, int kt, int nt, int ntn) {
  return ((const v16h*)(packed + (size_t)(kt * ntn + nt) * 512))[threadIdx.x & 31];
}
__device__ __forceinline__ v8f wmma32(v16h a, v16h b, v8f c) {
  return __builtin_amdgcn_wmma_f32_16x16x32_f16(false, a, false, b,
                                                (short)0, c, false, false);
}

// pack B (KxN f32 row-major) into fragment-ordered f16; grid (K/32, N/16), 32 thr
__global__ void k_packB(const float* __restrict__ B, int ldb, _Float16* __restrict__ o) {
  int lane = threadIdx.x, kt = blockIdx.x, nt = blockIdx.y;
  int n = nt * 16 + (lane & 15);
  int koff = (lane >> 4) * 8;
  int k0 = kt * 32;
  _Float16* dst = o + (size_t)(kt * gridDim.y + nt) * 512 + (size_t)lane * 16;
#pragma unroll
  for (int p = 0; p < 8; ++p) {
    int k = k0 + ((p < 4) ? 2 * p : 2 * p + 8) + koff;
    dst[2 * p]     = (_Float16)B[(size_t)k * ldb + n];
    dst[2 * p + 1] = (_Float16)B[(size_t)(k + 1) * ldb + n];
  }
}
// pack B^T source (Bt row-major NxK; B[k][n]=Bt[n][k]); grid (K/32, N/16)
__global__ void k_packBT(const float* __restrict__ Bt, int ldbt, _Float16* __restrict__ o) {
  int lane = threadIdx.x, kt = blockIdx.x, nt = blockIdx.y;
  int n = nt * 16 + (lane & 15);
  int koff = (lane >> 4) * 8;
  const float* row = Bt + (size_t)n * ldbt + kt * 32;
  _Float16* dst = o + (size_t)(kt * gridDim.y + nt) * 512 + (size_t)lane * 16;
#pragma unroll
  for (int p = 0; p < 8; ++p) {
    int k = ((p < 4) ? 2 * p : 2 * p + 8) + koff;
    dst[2 * p]     = (_Float16)row[k];
    dst[2 * p + 1] = (_Float16)row[k + 1];
  }
}

// ---------- precompute: l2norm(rel_emb) rows ----------
__global__ void k_relnorm(const float* __restrict__ rel, float* __restrict__ rnorm) {
  int r = blockIdx.x, lane = threadIdx.x;
  float4 v = ((const float4*)(rel + (size_t)r * FF))[lane];
  float ss = wsum(v.x * v.x + v.y * v.y + v.z * v.z + v.w * v.w);
  float inv = 1.0f / fmaxf(sqrtf(ss), EPSV);
  float4 o; o.x = v.x * inv; o.y = v.y * inv; o.z = v.z * inv; o.w = v.w * inv;
  ((float4*)(rnorm + (size_t)r * FF))[lane] = o;
}

// per-relation attention logit for each (dual, layer) combo
__global__ void k_attrel(const float* __restrict__ rnorm,
                         const float* __restrict__ attn_e,
                         const float* __restrict__ attn_r,
                         float* __restrict__ attRel) {
  int r = blockIdx.x, combo = blockIdx.y, lane = threadIdx.x;
  const float* attn = ((combo >> 1) ? attn_r : attn_e) + (combo & 1) * FF;
  float4 a = ((const float4*)(rnorm + (size_t)r * FF))[lane];
  float4 w = ((const float4*)attn)[lane];
  float d = wsum(a.x * w.x + a.y * w.y + a.z * w.z + a.w * w.w);
  if (lane == 0) attRel[(size_t)combo * NR + r] = d;
}

// ---------- initial features: mean of neighbor ent/rel embeddings ----------
__global__ void k_init_accum(const float* __restrict__ ent, const float* __restrict__ rel,
                             const int* __restrict__ esrc, const int* __restrict__ edst,
                             const int* __restrict__ erel,
                             float* deg, float* featE, float* featR) {
  int t = blockIdx.x * blockDim.x + threadIdx.x;
  int e = t >> 5, lane = t & 31;
  if (e >= NE) return;
  int s = esrc[e], d = edst[e], r = erel[e];
  if (lane == 0) atomicAdd(&deg[d], 1.0f);
  float4 ve = ((const float4*)(ent + (size_t)s * FF))[lane];
  float4 vr = ((const float4*)(rel + (size_t)r * FF))[lane];
  float* pe = featE + (size_t)d * FF + lane * 4;
  float* pr = featR + (size_t)d * FF + lane * 4;
  atomicAdd(pe + 0, ve.x); atomicAdd(pe + 1, ve.y);
  atomicAdd(pe + 2, ve.z); atomicAdd(pe + 3, ve.w);
  atomicAdd(pr + 0, vr.x); atomicAdd(pr + 1, vr.y);
  atomicAdd(pr + 2, vr.z); atomicAdd(pr + 3, vr.w);
}

__global__ void k_feat0(const float* __restrict__ deg, float* featE, float* featR,
                        float* outE, float* outR) {
  int n = blockIdx.x, lane = threadIdx.x;
  float dg = fmaxf(deg[n], 1.0f);
  size_t b = (size_t)n * FF + lane * 4;
  float4 e = *(float4*)(featE + b), r = *(float4*)(featR + b);
  e.x = tanhf(e.x / dg); e.y = tanhf(e.y / dg); e.z = tanhf(e.z / dg); e.w = tanhf(e.w / dg);
  r.x = tanhf(r.x / dg); r.y = tanhf(r.y / dg); r.z = tanhf(r.z / dg); r.w = tanhf(r.w / dg);
  *(float4*)(featE + b) = e; *(float4*)(featR + b) = r;
  *(float4*)(outE + (size_t)n * FD + lane * 4) = e;
  *(float4*)(outR + (size_t)n * FD + lane * 4) = r;
}

// ---------- edge softmax ----------
__global__ void k_edge_max(const int* __restrict__ edst, const int* __restrict__ erel,
                           const float* __restrict__ ar, unsigned* __restrict__ segmax) {
  int e = blockIdx.x * blockDim.x + threadIdx.x;
  if (e >= NE) return;
  atomicMax(&segmax[edst[e]], encf(ar[erel[e]]));
}
__global__ void k_edge_exp(const int* __restrict__ edst, const int* __restrict__ erel,
                           const float* __restrict__ ar, const unsigned* __restrict__ segmax,
                           float* __restrict__ edge_ex, float* __restrict__ segsum) {
  int e = blockIdx.x * blockDim.x + threadIdx.x;
  if (e >= NE) return;
  int d = edst[e];
  float ex = expf(ar[erel[e]] - decf(segmax[d]));
  edge_ex[e] = ex;
  atomicAdd(&segsum[d], ex);
}

// ---------- Householder-reflect + weighted scatter (one wave32 per edge) ----------
__global__ void k_edge_agg(const int* __restrict__ esrc, const int* __restrict__ edst,
                           const int* __restrict__ erel,
                           const float* __restrict__ feats, const float* __restrict__ rnorm,
                           const float* __restrict__ edge_ex, const float* __restrict__ segsum,
                           float* __restrict__ agg) {
  int t = blockIdx.x * blockDim.x + threadIdx.x;
  int e = t >> 5, lane = t & 31;
  if (e >= NE) return;
  int s = esrc[e], d = edst[e], r = erel[e];
  float4 nv = ((const float4*)(feats + (size_t)s * FF))[lane];
  float4 rv = ((const float4*)(rnorm + (size_t)r * FF))[lane];
  float dot = wsum(nv.x * rv.x + nv.y * rv.y + nv.z * rv.z + nv.w * rv.w);
  float w = edge_ex[e] / segsum[d];
  float c = 2.0f * dot;
  float* p = agg + (size_t)d * FF + lane * 4;
  atomicAdd(p + 0, (nv.x - c * rv.x) * w);
  atomicAdd(p + 1, (nv.y - c * rv.y) * w);
  atomicAdd(p + 2, (nv.z - c * rv.z) * w);
  atomicAdd(p + 3, (nv.w - c * rv.w) * w);
}

__global__ void k_layer_fin(const float* __restrict__ agg, float* __restrict__ feats,
                            float* __restrict__ ob, int coloff) {
  int n = blockIdx.x, lane = threadIdx.x;
  size_t b = (size_t)n * FF + lane * 4;
  float4 v = *(const float4*)(agg + b);
  v.x = tanhf(v.x); v.y = tanhf(v.y); v.z = tanhf(v.z); v.w = tanhf(v.w);
  *(float4*)(feats + b) = v;
  *(float4*)(ob + (size_t)n * FD + coloff + lane * 4) = v;
}

// ---------- proxy / gate phase ----------
__global__ void k_invnorm(const float* __restrict__ ob, float* __restrict__ invn) {
  int n = blockIdx.x, lane = threadIdx.x;
  const float4* row = (const float4*)(ob + (size_t)n * FD);
  float ss = 0.f;
#pragma unroll
  for (int i = 0; i < 3; ++i) {
    float4 v = row[lane + 32 * i];
    ss += v.x * v.x + v.y * v.y + v.z * v.z + v.w * v.w;
  }
  ss = wsum(ss);
  if (lane == 0) invn[n] = 1.0f / fmaxf(sqrtf(ss), EPSV);
}
__global__ void k_pnorm(const float* __restrict__ proxy, float* __restrict__ pnorm) {
  int p = blockIdx.x, lane = threadIdx.x;
  const float4* row = (const float4*)(proxy + (size_t)p * FD);
  float ss = 0.f;
  float4 v[3];
#pragma unroll
  for (int i = 0; i < 3; ++i) {
    v[i] = row[lane + 32 * i];
    ss += v[i].x * v[i].x + v[i].y * v[i].y + v[i].z * v[i].z + v[i].w * v[i].w;
  }
  ss = wsum(ss);
  float inv = 1.0f / fmaxf(sqrtf(ss), EPSV);
  float4* dst = (float4*)(pnorm + (size_t)p * FD);
#pragma unroll
  for (int i = 0; i < 3; ++i) {
    float4 o; o.x = v[i].x * inv; o.y = v[i].y * inv; o.z = v[i].z * inv; o.w = v[i].w * inv;
    dst[lane + 32 * i] = o;
  }
}

// logits[N,64] = (out * invn) @ pnorm^T    (K=384, packed B)
__global__ void k_logits(const float* __restrict__ ob, const float* __restrict__ invn,
                         const _Float16* __restrict__ pkPT, float* __restrict__ logits) {
  int m0 = blockIdx.x * 16;
  v8f c[4] = {};
  for (int kt = 0; kt < FD / 32; ++kt) {
    v16h a = ldAs(ob, FD, m0, kt * 32, invn);
#pragma unroll
    for (int t = 0; t < 4; ++t)
      c[t] = wmma32(a, ldP(pkPT, kt, t, 4), c[t]);
  }
  int lane = threadIdx.x, hi = lane >> 4, nn = lane & 15;
#pragma unroll
  for (int t = 0; t < 4; ++t)
#pragma unroll
    for (int v = 0; v < 8; ++v)
      logits[(size_t)(m0 + v + 8 * hi) * PP + t * 16 + nn] = c[t][v];
}

__global__ void k_softmax64(float* __restrict__ logits) {
  int n = blockIdx.x, lane = threadIdx.x;
  float a0 = logits[(size_t)n * PP + lane];
  float a1 = logits[(size_t)n * PP + 32 + lane];
  float m = wmaxr(fmaxf(a0, a1));
  float e0 = expf(a0 - m), e1 = expf(a1 - m);
  float s = wsum(e0 + e1);
  logits[(size_t)n * PP + lane] = e0 / s;
  logits[(size_t)n * PP + 32 + lane] = e1 / s;
}

// pf = out - att @ proxy     (K=64, packed B with 24 n-tiles)
__global__ void k_pf(const float* __restrict__ att, const _Float16* __restrict__ pkProxy,
                     const float* __restrict__ ob, float* __restrict__ pf) {
  int m0 = blockIdx.x * 16, n0 = blockIdx.y * 64;
  v8f c[4] = {};
  for (int kt = 0; kt < PP / 32; ++kt) {
    v16h a = ldA(att, PP, m0, kt * 32);
#pragma unroll
    for (int t = 0; t < 4; ++t)
      c[t] = wmma32(a, ldP(pkProxy, kt, blockIdx.y * 4 + t, FD / 16), c[t]);
  }
  int lane = threadIdx.x, hi = lane >> 4, nn = lane & 15;
#pragma unroll
  for (int t = 0; t < 4; ++t)
#pragma unroll
    for (int v = 0; v < 8; ++v) {
      size_t idx = (size_t)(m0 + v + 8 * hi) * FD + n0 + t * 16 + nn;
      pf[idx] = ob[idx] - c[t][v];
    }
}

// gate = sigmoid(pf @ G + b); d_out[:, dualoff+n] = gate*out + (1-gate)*pf   (K=384)
__global__ void k_gate(const float* __restrict__ pf, const _Float16* __restrict__ pkG,
                       const float* __restrict__ bias, const float* __restrict__ ob,
                       float* __restrict__ dout, int dualoff) {
  int m0 = blockIdx.x * 16, n0 = blockIdx.y * 64;
  v8f c[4] = {};
  for (int kt = 0; kt < FD / 32; ++kt) {
    v16h a = ldA(pf, FD, m0, kt * 32);
#pragma unroll
    for (int t = 0; t < 4; ++t)
      c[t] = wmma32(a, ldP(pkG, kt, blockIdx.y * 4 + t, FD / 16), c[t]);
  }
  int lane = threadIdx.x, hi = lane >> 4, nn = lane & 15;
#pragma unroll
  for (int t = 0; t < 4; ++t)
#pragma unroll
    for (int v = 0; v < 8; ++v) {
      int m = m0 + v + 8 * hi, n = n0 + t * 16 + nn;
      float g = 1.0f / (1.0f + expf(-(c[t][v] + bias[n])));
      size_t idx = (size_t)m * FD + n;
      dout[(size_t)m * (2 * FD) + dualoff + n] = g * ob[idx] + (1.0f - g) * pf[idx];
    }
}

extern "C" void kernel_launch(void* const* d_in, const int* in_sizes, int n_in,
                              void* d_out, int out_size, void* d_ws, size_t ws_size,
                              hipStream_t stream) {
  (void)in_sizes; (void)n_in; (void)out_size; (void)ws_size;
  const float* ent     = (const float*)d_in[0];
  const float* rel     = (const float*)d_in[1];
  const int*   esrc    = (const int*)d_in[2];
  const int*   edst    = (const int*)d_in[3];
  const int*   erel    = (const int*)d_in[4];
  const float* attn_e  = (const float*)d_in[5];
  const float* gate_e  = (const float*)d_in[6];
  const float* proxy_e = (const float*)d_in[7];
  const float* bias_e  = (const float*)d_in[8];
  const float* attn_r  = (const float*)d_in[9];
  const float* gate_r  = (const float*)d_in[10];
  const float* proxy_r = (const float*)d_in[11];
  const float* bias_r  = (const float*)d_in[12];
  float* out = (float*)d_out;

  float* ws = (float*)d_ws;
  size_t off = 0;
  auto alloc = [&](size_t nfloats) {
    off = (off + 7) & ~(size_t)7;     // 32B alignment for v16h loads
    float* p = ws + off; off += nfloats; return p;
  };
  float*    deg     = alloc(NN);
  float*    featE   = alloc((size_t)NN * FF);
  float*    featR   = alloc((size_t)NN * FF);
  float*    agg     = alloc((size_t)NN * FF);
  float*    rnorm   = alloc((size_t)NR * FF);
  float*    attRel  = alloc(4 * NR);
  unsigned* segmax  = (unsigned*)alloc(NN);
  float*    segsum  = alloc(NN);
  float*    edge_ex = alloc(NE);
  float*    outE    = alloc((size_t)NN * FD);
  float*    outR    = alloc((size_t)NN * FD);
  float*    pf      = alloc((size_t)NN * FD);
  float*    invn    = alloc(NN);
  float*    pnorm   = alloc((size_t)PP * FD);
  float*    logits  = alloc((size_t)NN * PP);
  _Float16* pkG     = (_Float16*)alloc((size_t)FD * FD / 2);   // 384x384 halves
  _Float16* pkProxy = (_Float16*)alloc((size_t)PP * FD / 2);   // 64x384 halves
  _Float16* pkPT    = (_Float16*)alloc((size_t)FD * PP / 2);   // 384x64 halves

  hipMemsetAsync(deg, 0, NN * sizeof(float), stream);
  hipMemsetAsync(featE, 0, (size_t)NN * FF * sizeof(float), stream);
  hipMemsetAsync(featR, 0, (size_t)NN * FF * sizeof(float), stream);

  k_relnorm<<<NR, 32, 0, stream>>>(rel, rnorm);
  k_attrel<<<dim3(NR, 4), 32, 0, stream>>>(rnorm, attn_e, attn_r, attRel);
  k_init_accum<<<NE / 8, 256, 0, stream>>>(ent, rel, esrc, edst, erel, deg, featE, featR);
  k_feat0<<<NN, 32, 0, stream>>>(deg, featE, featR, outE, outR);

  for (int dual = 0; dual < 2; ++dual) {
    float* feats = dual ? featR : featE;
    float* ob    = dual ? outR : outE;
    for (int layer = 0; layer < 2; ++layer) {
      const float* ar = attRel + (size_t)(dual * 2 + layer) * NR;
      hipMemsetAsync(segmax, 0, NN * sizeof(unsigned), stream);
      hipMemsetAsync(segsum, 0, NN * sizeof(float), stream);
      hipMemsetAsync(agg, 0, (size_t)NN * FF * sizeof(float), stream);
      k_edge_max<<<(NE + 255) / 256, 256, 0, stream>>>(edst, erel, ar, segmax);
      k_edge_exp<<<(NE + 255) / 256, 256, 0, stream>>>(edst, erel, ar, segmax, edge_ex, segsum);
      k_edge_agg<<<NE / 8, 256, 0, stream>>>(esrc, edst, erel, feats, rnorm, edge_ex, segsum, agg);
      k_layer_fin<<<NN, 32, 0, stream>>>(agg, feats, ob, (layer + 1) * FF);
    }
    const float* proxy = dual ? proxy_r : proxy_e;
    const float* gw    = dual ? gate_r : gate_e;
    const float* bias  = dual ? bias_r : bias_e;
    k_invnorm<<<NN, 32, 0, stream>>>(ob, invn);
    k_pnorm<<<PP, 32, 0, stream>>>(proxy, pnorm);
    // pack static B operands into WMMA fragment order (f16, once per dual)
    k_packB <<<dim3(FD / 32, FD / 16), 32, 0, stream>>>(gw, FD, pkG);
    k_packB <<<dim3(PP / 32, FD / 16), 32, 0, stream>>>(proxy, FD, pkProxy);
    k_packBT<<<dim3(FD / 32, PP / 16), 32, 0, stream>>>(pnorm, FD, pkPT);
    k_logits<<<NN / 16, 32, 0, stream>>>(ob, invn, pkPT, logits);
    k_softmax64<<<NN, 32, 0, stream>>>(logits);
    k_pf<<<dim3(NN / 16, FD / 64), 32, 0, stream>>>(logits, pkProxy, ob, pf);
    k_gate<<<dim3(NN / 16, FD / 64), 32, 0, stream>>>(pf, pkG, bias, ob, out, dual * FD);
  }
}
